// MeshAutoencoder_54219667144824
// MI455X (gfx1250) — matrix operand order repaired and checked by reading, not compiled, not run on hardware
//
#include <hip/hip_runtime.h>
#include <hip/hip_bf16.h>

typedef __attribute__((ext_vector_type(16))) _Float16 v16h;
typedef __attribute__((ext_vector_type(8)))  float    v8f;

#define NN   131072     // b*nf nodes
#define NF_  16384
#define NV_  8192
#define BDIM 8
#define EPB  49152
#define EE   (BDIM*EPB)

#define GF_BIAS 1
#define GF_RELU 2

// ---------------- elementwise helpers ----------------

// x f32 [NN, din] -> second half of acat rows: acat[n*2din + din + j]
__global__ void k_f32_to_f16_strided(const float* __restrict__ s, _Float16* __restrict__ acat, int din) {
    int i = blockIdx.x * 256 + threadIdx.x;
    if (i >= NN * din) return;
    int n = i / din, j = i - n * din;
    acat[(size_t)n * (2 * din) + din + j] = (_Float16)s[i];
}

__global__ void k_zero(float* __restrict__ p, int n) {
    int i = blockIdx.x * 256 + threadIdx.x;
    if (i < n) p[i] = 0.0f;
}

// Convert f32 weight(s) [K,N] row-major -> f16 in WMMA-fragment-ready layout:
// block (bk = k/32, bn = n/16): 32 lanes x 16 halves; lane l owns col bn*16+(l&15),
// k = bk*32 + (l>>4)*16 + j  (j = 0..15 contiguous).  Dual-source builds [Wl;Wr].
__global__ void k_cvt_swizzle(const float* __restrict__ srcL, const float* __restrict__ srcR,
                              int split, _Float16* __restrict__ out, int K, int N) {
    int idx = blockIdx.x * 256 + threadIdx.x;
    if (idx >= K * N) return;
    int j   = idx & 15;
    int l   = (idx >> 4) & 31;
    int blk = idx >> 9;
    int nb  = N >> 4;
    int bn  = blk % nb, bk = blk / nb;
    int k   = bk * 32 + (l >> 4) * 16 + j;
    int col = bn * 16 + (l & 15);
    float v = (k < split) ? srcL[(size_t)k * N + col]
                          : srcR[(size_t)(k - split) * N + col];
    out[idx] = (_Float16)v;
}

// Build A matrix [NN, 576] f16: per-face concatenated coordinate embeddings
__global__ void k_embed(const float* __restrict__ vertices, const int* __restrict__ faces,
                        const float* __restrict__ coor_embed, _Float16* __restrict__ out) {
    int idx = blockIdx.x * 256 + threadIdx.x;
    if (idx >= NN * 576) return;
    int n = idx / 576, j = idx - n * 576;
    int b = n / NF_;
    int f0 = faces[n * 3 + 0], f1 = faces[n * 3 + 1], f2 = faces[n * 3 + 2];
    bool m = (f0 != -1) && (f1 != -1) && (f2 != -1);
    int e9 = j >> 6;          // 0..8  (vertex*3 + coord)
    int dd = j & 63;          // embed dim
    int vsel = e9 / 3;
    int c = e9 - vsel * 3;
    int vi = m ? (vsel == 0 ? f0 : (vsel == 1 ? f1 : f2)) : 0;
    float val = vertices[((size_t)b * NV_ + (size_t)vi) * 3 + c];
    float t = (val + 1.0f) * 0.5f * 128.0f - 0.5f;
    int disc = (int)rintf(t);
    disc = min(127, max(0, disc));
    out[idx] = (_Float16)coor_embed[disc * 64 + dd];
}

// ---------------- WMMA GEMM: C[M,N] = A[M,K](f16,lda) * Bsw[K,N](f16 pre-swizzled) ----------------
// block tile 128x64, 8 waves, wave tile 32x32 (2x2 frags), K-step 32.
// A: register-staged double-buffered LDS (frags = 2x ds_load_b128 each).
// B: fragment-ready in global memory, 2x global_load_b128 per frag (L2 resident).

template <int FLAGS>
__global__ __launch_bounds__(256)
void k_gemm(const _Float16* __restrict__ A, int lda, const _Float16* __restrict__ Bsw,
            const float* __restrict__ bias, float* __restrict__ C,
            int M, int N, int K) {
    __shared__ _Float16 As[2][128][40];   // 32 + 8 pad halves per row

    union V16 { v16h v; uint4 q[2]; };

    const int tid  = threadIdx.x;
    const int lane = tid & 31;
    const int wave = tid >> 5;
    const int wm = wave & 3;            // 4 waves along M
    const int wn = wave >> 2;           // 2 waves along N
    const int hi = lane >> 4;
    const int lr = lane & 15;
    const int m0 = blockIdx.x * 128;
    const int n0 = blockIdx.y * 64;

    const int ar0 = tid >> 2, ac0 = (tid & 3) * 8;   // A tile: 2 uint4/thread
    const int nb16 = N >> 4;
    const size_t bfrag0 = ((size_t)(n0 >> 4) + wn * 2) * 512 + lane * 16;

    v8f acc[2][2];
    v8f vzero = {0.f, 0.f, 0.f, 0.f, 0.f, 0.f, 0.f, 0.f};
#pragma unroll
    for (int i = 0; i < 2; ++i)
#pragma unroll
        for (int j = 0; j < 2; ++j) acc[i][j] = vzero;

    auto gloadA = [&](int k0, uint4& a0, uint4& a1) {
        a0 = *(const uint4*)&A[(size_t)(m0 + ar0) * lda + k0 + ac0];
        a1 = *(const uint4*)&A[(size_t)(m0 + ar0 + 64) * lda + k0 + ac0];
    };

    uint4 a0, a1;
    gloadA(0, a0, a1);
    const int nIter = K >> 5;

    for (int it = 0; it < nIter; ++it) {
        const int buf = it & 1;
        *(uint4*)&As[buf][ar0][ac0]      = a0;
        *(uint4*)&As[buf][ar0 + 64][ac0] = a1;
        __syncthreads();
        if (it + 1 < nIter) gloadA((it + 1) << 5, a0, a1);   // overlap with compute

        // B frags straight from pre-swizzled global (coalesced b128, L2-resident)
        V16 bf[2];
        const size_t bb = (size_t)it * nb16 * 512 + bfrag0;
#pragma unroll
        for (int ni = 0; ni < 2; ++ni) {
            bf[ni].q[0] = *(const uint4*)&Bsw[bb + ni * 512];
            bf[ni].q[1] = *(const uint4*)&Bsw[bb + ni * 512 + 8];
        }
        // A frags: one contiguous 16-half run, k = hi*16 + j (matches swizzled B)
        V16 af[2];
#pragma unroll
        for (int mi = 0; mi < 2; ++mi) {
            int row = wm * 32 + mi * 16 + lr;
            af[mi].q[0] = *(const uint4*)&As[buf][row][hi * 16];
            af[mi].q[1] = *(const uint4*)&As[buf][row][hi * 16 + 8];
        }
#pragma unroll
        for (int mi = 0; mi < 2; ++mi)
#pragma unroll
            for (int ni = 0; ni < 2; ++ni)
                acc[mi][ni] = __builtin_amdgcn_wmma_f32_16x16x32_f16(
                    false, af[mi].v, false, bf[ni].v, (short)0, acc[mi][ni], false, false);
    }

    // epilogue: C/D layout — VGPR r, lanes0-15 M=r, lanes16-31 M=r+8, N=lr
#pragma unroll
    for (int mi = 0; mi < 2; ++mi) {
#pragma unroll
        for (int ni = 0; ni < 2; ++ni) {
            int gn = n0 + wn * 32 + ni * 16 + lr;
            float bv = (FLAGS & GF_BIAS) ? bias[gn] : 0.0f;
#pragma unroll
            for (int r = 0; r < 8; ++r) {
                int gm = m0 + wm * 32 + mi * 16 + hi * 8 + r;
                float v = acc[mi][ni][r] + bv;
                if (FLAGS & GF_RELU) v = fmaxf(v, 0.0f);
                C[(size_t)gm * N + gn] = v;
            }
        }
    }
}

// ---------------- graph aggregation: wave per edge, fp32 atomics ----------------

__global__ __launch_bounds__(256)
void k_edge_agg(const float* __restrict__ xp, const int* __restrict__ fe,
                float* __restrict__ agg, float* __restrict__ cnt, int din) {
    int wave = (blockIdx.x * 256 + threadIdx.x) >> 5;
    int lane = threadIdx.x & 31;
    if (wave >= EE) return;
    int bidx = wave / EPB;
    int s = fe[(size_t)wave * 2];
    int d = fe[(size_t)wave * 2 + 1];
    if (s < 0 || d < 0) return;                 // padded edge (weight 0)
    size_t src = (size_t)bidx * NF_ + (size_t)s;
    size_t dst = (size_t)bidx * NF_ + (size_t)d;
    for (int j = lane; j < din; j += 32)
        atomicAdd(&agg[dst * din + j], xp[src * din + j]);
    if (lane == 0) atomicAdd(&cnt[dst], 1.0f);
}

// mean -> f16 into FIRST half of acat rows (stride 2*din)
__global__ void k_mean_h(const float* __restrict__ agg, const float* __restrict__ cnt,
                         _Float16* __restrict__ acat, int din) {
    int i = blockIdx.x * 256 + threadIdx.x;
    if (i >= NN * din) return;
    int n = i / din, j = i - n * din;
    float c = fmaxf(cnt[n], 1.0f);
    acat[(size_t)n * (2 * din) + j] = (_Float16)(agg[i] / c);
}

// ---------------- row L2 normalize (wave per row) ----------------

__global__ __launch_bounds__(256)
void k_l2norm(const float* __restrict__ src, float* __restrict__ dst, int d) {
    int row  = (blockIdx.x * 256 + threadIdx.x) >> 5;
    int lane = threadIdx.x & 31;
    if (row >= NN) return;
    const float* p = src + (size_t)row * d;
    float ss = 0.0f;
    for (int j = lane; j < d; j += 32) { float v = p[j]; ss += v * v; }
#pragma unroll
    for (int o = 16; o > 0; o >>= 1) ss += __shfl_xor(ss, o, 32);
    float rn = 1.0f / fmaxf(sqrtf(ss), 1e-12f);
    float* q = dst + (size_t)row * d;
    for (int j = lane; j < d; j += 32) q[j] = p[j] * rn;
}

// final layer: in-place normalize + face mask on d_out (d = 576)
__global__ __launch_bounds__(256)
void k_l2norm_mask(float* __restrict__ x, const int* __restrict__ faces, int d) {
    int row  = (blockIdx.x * 256 + threadIdx.x) >> 5;
    int lane = threadIdx.x & 31;
    if (row >= NN) return;
    int f0 = faces[row * 3 + 0], f1 = faces[row * 3 + 1], f2 = faces[row * 3 + 2];
    float mask = ((f0 != -1) && (f1 != -1) && (f2 != -1)) ? 1.0f : 0.0f;
    float* p = x + (size_t)row * d;
    float ss = 0.0f;
    for (int j = lane; j < d; j += 32) { float v = p[j]; ss += v * v; }
#pragma unroll
    for (int o = 16; o > 0; o >>= 1) ss += __shfl_xor(ss, o, 32);
    float rn = mask / fmaxf(sqrtf(ss), 1e-12f);
    for (int j = lane; j < d; j += 32) p[j] = p[j] * rn;
}

// silu + layernorm(64) in place, wave per row, 2 elems/lane
__global__ __launch_bounds__(256)
void k_silu_ln(float* __restrict__ x, const float* __restrict__ g, const float* __restrict__ b) {
    int row  = (blockIdx.x * 256 + threadIdx.x) >> 5;
    int lane = threadIdx.x & 31;
    if (row >= NN) return;
    float* p = x + (size_t)row * 64;
    float v0 = p[lane], v1 = p[lane + 32];
    v0 = v0 / (1.0f + __expf(-v0));
    v1 = v1 / (1.0f + __expf(-v1));
    float s = v0 + v1;
#pragma unroll
    for (int o = 16; o > 0; o >>= 1) s += __shfl_xor(s, o, 32);
    float mu = s * (1.0f / 64.0f);
    float d0 = v0 - mu, d1 = v1 - mu;
    float q = d0 * d0 + d1 * d1;
#pragma unroll
    for (int o = 16; o > 0; o >>= 1) q += __shfl_xor(q, o, 32);
    float rs = rsqrtf(q * (1.0f / 64.0f) + 1e-5f);
    p[lane]      = d0 * rs * g[lane]      + b[lane];
    p[lane + 32] = d1 * rs * g[lane + 32] + b[lane + 32];
}

// ---------------- host orchestration ----------------

extern "C" void kernel_launch(void* const* d_in, const int* in_sizes, int n_in,
                              void* d_out, int out_size, void* d_ws, size_t ws_size,
                              hipStream_t stream) {
    (void)in_sizes; (void)n_in; (void)out_size; (void)ws_size;
    const float* vertices   = (const float*)d_in[0];
    const int*   faces      = (const int*)d_in[1];
    const int*   face_edges = (const int*)d_in[2];
    const float* coor_embed = (const float*)d_in[3];
    const float* proj_W     = (const float*)d_in[4];
    const float* proj_b     = (const float*)d_in[5];
    const float* ln_g       = (const float*)d_in[6];
    const float* ln_b       = (const float*)d_in[7];
    const float *Wp[5], *bp[5], *Wl[5], *bl[5], *Wr[5];
    for (int i = 0; i < 5; ++i) {
        Wp[i] = (const float*)d_in[8 + 5 * i + 0];
        bp[i] = (const float*)d_in[8 + 5 * i + 1];
        Wl[i] = (const float*)d_in[8 + 5 * i + 2];
        bl[i] = (const float*)d_in[8 + 5 * i + 3];
        Wr[i] = (const float*)d_in[8 + 5 * i + 4];
    }
    static const int DIN[5]  = {192, 64, 128, 256, 256};
    static const int DOUT[5] = {64, 128, 256, 256, 576};

    char* base = (char*)d_ws;
    size_t off = 0;
    auto alloc = [&](size_t bytes) -> void* {
        void* p = base + off; off += (bytes + 255) & ~(size_t)255; return p;
    };
    _Float16* xh   = (_Float16*)alloc((size_t)NN * 576 * 2);  // embed A [NN,576]
    _Float16* acat = (_Float16*)alloc((size_t)NN * 512 * 2);  // [mean | x] f16, stride 2*din
    float*    x    = (float*)   alloc((size_t)NN * 256 * 4);  // f32 activation
    float*    xp   = (float*)   alloc((size_t)NN * 256 * 4);  // relu-projected
    float*    agg  = (float*)   alloc((size_t)NN * 256 * 4);
    float*    cnt  = (float*)   alloc((size_t)NN * 4);
    _Float16* projWH = (_Float16*)alloc((size_t)576 * 192 * 2);   // swizzled
    _Float16 *WpH[5], *WlrH[5];
    for (int i = 0; i < 5; ++i) {
        WpH[i]  = (_Float16*)alloc((size_t)DIN[i] * DIN[i] * 2);           // swizzled
        WlrH[i] = (_Float16*)alloc((size_t)2 * DIN[i] * DOUT[i] * 2);      // swizzled [Wl ; Wr]
    }
    float* out = (float*)d_out;   // [NN, 576]; also used as pre-normalize GEMM scratch

    auto swz = [&](const float* sL, const float* sR, int split, _Float16* d, int K, int N) {
        int n = K * N;
        k_cvt_swizzle<<<(n + 255) / 256, 256, 0, stream>>>(sL, sR, split, d, K, N);
    };
    swz(proj_W, proj_W, 576, projWH, 576, 192);
    for (int i = 0; i < 5; ++i) {
        swz(Wp[i], Wp[i], DIN[i], WpH[i], DIN[i], DIN[i]);
        swz(Wl[i], Wr[i], DIN[i], WlrH[i], 2 * DIN[i], DOUT[i]);
    }

    // embedding -> A[NN,576] f16, then x0 = A @ projW + b  -> x [NN,192]
    k_embed<<<(NN * 576 + 255) / 256, 256, 0, stream>>>(vertices, faces, coor_embed, xh);
    k_gemm<GF_BIAS><<<dim3(NN / 128, 192 / 64), 256, 0, stream>>>(
        xh, 576, projWH, proj_b, x, NN, 192, 576);

    for (int L = 0; L < 5; ++L) {
        const int din = DIN[L], dout = DOUT[L], K2 = 2 * din;
        // x -> f16 into second half of acat rows
        k_f32_to_f16_strided<<<(NN * din + 255) / 256, 256, 0, stream>>>(x, acat, din);
        // xp = relu(x @ Wp + bp)   (A = x-half of acat, lda = 2*din)
        k_gemm<GF_BIAS | GF_RELU><<<dim3(NN / 128, din / 64), 256, 0, stream>>>(
            acat + din, K2, WpH[L], bp[L], xp, NN, din, din);
        // mean aggregation over edges
        k_zero<<<(NN * din + 255) / 256, 256, 0, stream>>>(agg, NN * din);
        k_zero<<<(NN + 255) / 256, 256, 0, stream>>>(cnt, NN);
        k_edge_agg<<<EE / 8, 256, 0, stream>>>(xp, face_edges, agg, cnt, din);
        k_mean_h<<<(NN * din + 255) / 256, 256, 0, stream>>>(agg, cnt, acat, din);
        // out = [mean|x] @ [Wl;Wr] + bl   (single fused GEMM, K = 2*din)
        k_gemm<GF_BIAS><<<dim3(NN / 128, dout / 64), 256, 0, stream>>>(
            acat, K2, WlrH[L], bl[L], out, NN, dout, K2);
        if (L < 4) {
            k_l2norm<<<NN / 8, 256, 0, stream>>>(out, x, dout);
            if (L == 0) k_silu_ln<<<NN / 8, 256, 0, stream>>>(x, ln_g, ln_b);
        } else {
            k_l2norm_mask<<<NN / 8, 256, 0, stream>>>(out, faces, dout);  // in place on d_out
        }
    }
}
